// LinearLIFBlock_13400297963915
// MI455X (gfx1250) — matrix operand order repaired and checked by reading, not compiled, not run on hardware
//
#include <hip/hip_runtime.h>
#include <stdint.h>

// ---------------------------------------------------------------------------
// LinearLIFBlock for MI455X (gfx1250, wave32, WMMA + async global->LDS DMA)
//   cur = x @ W^T  (bf16 WMMA, f32 accum)  ->  LIF scan over T (in-place)
// ---------------------------------------------------------------------------

typedef __bf16 bf16;
typedef __attribute__((ext_vector_type(16))) __bf16 v16bf;
typedef __attribute__((ext_vector_type(8)))  __bf16 v8bf;
typedef __attribute__((ext_vector_type(4)))  __bf16 v4bf;
typedef __attribute__((ext_vector_type(8)))  float  v8f;

#define T_STEPS 64
#define BATCH   256
#define FIN     1024
#define FOUT    1024
#define M_TOT   (T_STEPS * BATCH)   // 16384 GEMM rows

#define BM 128
#define BN 128
#define BK 32      // one v_wmma_f32_16x16x32_bf16 K-step per tile
#define KPAD 40    // padded LDS row stride (bf16 elems) -> conflict-free b128 reads

// --- CDNA5 async global->LDS path (inline asm issue + ASYNCcnt wait) -------
#if defined(__has_builtin)
#if __has_builtin(__builtin_amdgcn_s_wait_asynccnt)
#define USE_ASYNC_LDS 1
#endif
#endif
#ifndef USE_ASYNC_LDS
#define USE_ASYNC_LDS 0
#endif

#if USE_ASYNC_LDS
// GLOBAL_LOAD_ASYNC_TO_LDS_B128: each lane DMAs 16B global -> its LDS address.
// vdst = LDS byte address (low 32 bits of generic __shared__ pointer),
// vaddr = 64-bit global address, no SADDR ("off"). Tracked on ASYNCcnt.
__device__ __forceinline__ void async_copy_b128(const void* gsrc,
                                                const void* ldst) {
  const uint64_t ga = (uint64_t)(uintptr_t)gsrc;
  const uint32_t la = (uint32_t)(uintptr_t)ldst;
  asm volatile("global_load_async_to_lds_b128 %0, %1, off"
               :
               : "v"(la), "v"(ga)
               : "memory");
}
#endif

// ---------------------------------------------------------------------------
// Kernel 1: f32 -> bf16 pack (bandwidth-bound, one pass)
// ---------------------------------------------------------------------------
__global__ void cvt_f32_to_bf16(const float* __restrict__ src,
                                bf16* __restrict__ dst, int n4) {
  int i = blockIdx.x * blockDim.x + threadIdx.x;
  if (i >= n4) return;
  float4 f = reinterpret_cast<const float4*>(src)[i];
  v4bf b;
  b[0] = (bf16)f.x; b[1] = (bf16)f.y; b[2] = (bf16)f.z; b[3] = (bf16)f.w;
  reinterpret_cast<v4bf*>(dst)[i] = b;
}

// ---------------------------------------------------------------------------
// Kernel 2: GEMM  C[m,o] = sum_f A[m,f] * B[o,f]
//   A: [M_TOT][FIN] bf16, B: [FOUT][FIN] bf16 (K contiguous), C: [M_TOT][FOUT] f32
// 8 waves per block: wave (wm in 0..3, wn in 0..1) computes a 32x64 patch.
// ---------------------------------------------------------------------------
__global__ __launch_bounds__(256)
void gemm_bf16_wmma(const bf16* __restrict__ A,
                    const bf16* __restrict__ B,
                    float* __restrict__ C) {
  __shared__ bf16 As[2][BM * KPAD];   // 2 x 10 KB
  __shared__ bf16 Bs[2][BN * KPAD];   // 2 x 10 KB

  const int tid  = threadIdx.x;
  const int lane = tid & 31;
  const int wave = tid >> 5;
  const int wm   = wave & 3;   // row group  (4)
  const int wn   = wave >> 2;  // col group  (2)

  const int m0 = blockIdx.y * BM;
  const int n0 = blockIdx.x * BN;

  v8f acc[2][4];
#pragma unroll
  for (int i = 0; i < 2; ++i)
#pragma unroll
    for (int j = 0; j < 4; ++j) acc[i][j] = {};

  // Tile staging: 128 rows x 64 B = 512 x 16B chunks; 256 threads x 2 chunks.
  const int seg  = tid & 3;       // 16B segment within a 64B row
  const int row0 = tid >> 2;      // rows tid>>2 and tid>>2 + 64

#if USE_ASYNC_LDS
  // CDNA5: DMA each 16B chunk global->LDS on ASYNCcnt (no VGPR transit)
  auto stageTile = [&](const bf16* gbase, int rowBase, int k0, bf16* lds) {
#pragma unroll
    for (int j = 0; j < 2; ++j) {
      const int row = row0 + j * 64;
      const bf16* g = gbase + (size_t)(rowBase + row) * FIN + k0 + seg * 8;
      bf16* l = lds + row * KPAD + seg * 8;
      async_copy_b128(g, l);
    }
  };
#else
  auto loadGlobal = [&](const bf16* gbase, int rowBase, int k0, uint4* r) {
#pragma unroll
    for (int j = 0; j < 2; ++j) {
      const int row = row0 + j * 64;
      r[j] = *reinterpret_cast<const uint4*>(
          gbase + (size_t)(rowBase + row) * FIN + k0 + seg * 8);
    }
  };
  auto storeLds = [&](bf16* dst, const uint4* r) {
#pragma unroll
    for (int j = 0; j < 2; ++j) {
      const int row = row0 + j * 64;
      *reinterpret_cast<uint4*>(dst + row * KPAD + seg * 8) = r[j];
    }
  };
#endif

  const int kh = lane >> 4;   // K-half select per CDNA5 WMMA VGPR layout
  const int lr = lane & 15;
  const int NK = FIN / BK;    // 32 K-tiles

#if USE_ASYNC_LDS
  stageTile(A, m0, 0, As[0]);
  stageTile(B, n0, 0, Bs[0]);
  __builtin_amdgcn_s_wait_asynccnt(0);
  __syncthreads();
#else
  uint4 ra[2], rb[2];
  loadGlobal(A, m0, 0, ra);
  loadGlobal(B, n0, 0, rb);
  storeLds(As[0], ra);
  storeLds(Bs[0], rb);
  __syncthreads();
#endif

  for (int kt = 0; kt < NK; ++kt) {
    const int buf = kt & 1;

#if USE_ASYNC_LDS
    if (kt + 1 < NK) {   // DMA next tile into the buffer read last iteration
      stageTile(A, m0, (kt + 1) * BK, As[buf ^ 1]);
      stageTile(B, n0, (kt + 1) * BK, Bs[buf ^ 1]);
    }
#else
    if (kt + 1 < NK) {
      loadGlobal(A, m0, (kt + 1) * BK, ra);
      loadGlobal(B, n0, (kt + 1) * BK, rb);
    }
#endif

    // A frag (16x32 bf16): lane<16 -> K {0..7,16..23}; lane>=16 -> K {8..15,24..31}
    v16bf afrag[2];
#pragma unroll
    for (int ms = 0; ms < 2; ++ms) {
      const bf16* p = &As[buf][(wm * 32 + ms * 16 + lr) * KPAD + kh * 8];
      v8bf lo = *reinterpret_cast<const v8bf*>(p);
      v8bf hi = *reinterpret_cast<const v8bf*>(p + 16);
      afrag[ms] = __builtin_shufflevector(lo, hi, 0, 1, 2, 3, 4, 5, 6, 7,
                                          8, 9, 10, 11, 12, 13, 14, 15);
    }
    // B frag (32x16 bf16): lane<16 -> col lr, K {0..7,16..23}; lane>=16 -> K {8..15,24..31}
    v16bf bfrag[4];
#pragma unroll
    for (int ns = 0; ns < 4; ++ns) {
      const bf16* p = &Bs[buf][(wn * 64 + ns * 16 + lr) * KPAD + kh * 8];
      v8bf lo = *reinterpret_cast<const v8bf*>(p);
      v8bf hi = *reinterpret_cast<const v8bf*>(p + 16);
      bfrag[ns] = __builtin_shufflevector(lo, hi, 0, 1, 2, 3, 4, 5, 6, 7,
                                          8, 9, 10, 11, 12, 13, 14, 15);
    }

#pragma unroll
    for (int ms = 0; ms < 2; ++ms)
#pragma unroll
      for (int ns = 0; ns < 4; ++ns)
        acc[ms][ns] = __builtin_amdgcn_wmma_f32_16x16x32_bf16(
            false, afrag[ms], false, bfrag[ns], (short)0, acc[ms][ns],
            false, false);

#if USE_ASYNC_LDS
    __builtin_amdgcn_s_wait_asynccnt(0);   // our DMAs into LDS[buf^1] landed
    __syncthreads();                       // everyone done reading LDS[buf]
#else
    __syncthreads();
    if (kt + 1 < NK) {
      storeLds(As[buf ^ 1], ra);
      storeLds(Bs[buf ^ 1], rb);
    }
    __syncthreads();
#endif
  }

  // Epilogue: C/D layout -> VGPR i: lanes 0-15 M=i, lanes 16-31 M=8+i; N=lane%16
  const int rowAdd = (lane >> 4) * 8;
#pragma unroll
  for (int ms = 0; ms < 2; ++ms) {
#pragma unroll
    for (int ns = 0; ns < 4; ++ns) {
      const int col  = n0 + wn * 64 + ns * 16 + lr;
      const int rowb = m0 + wm * 32 + ms * 16 + rowAdd;
#pragma unroll
      for (int i = 0; i < 8; ++i)
        C[(size_t)(rowb + i) * FOUT + col] = acc[ms][ns][i];
    }
  }
}

// ---------------------------------------------------------------------------
// Kernel 3: LIF scan over T, in place on cur (d_out), float4 per thread.
//   h = v + (x - v)/2 ; s = (h >= 1) ; v = h * (1 - s)
// ---------------------------------------------------------------------------
__device__ __forceinline__ void lif_step(float& v, float x, float& s) {
  const float h = v + (x - v) * 0.5f;
  s = (h >= 1.0f) ? 1.0f : 0.0f;
  v = h * (1.0f - s);   // hard reset
}

__global__ void lif_scan_kernel(float* __restrict__ cur_spk) {
  const int idx4 = blockIdx.x * blockDim.x + threadIdx.x;  // 0 .. B*FOUT/4-1
  const int BF4  = (BATCH * FOUT) / 4;
  float4* p = reinterpret_cast<float4*>(cur_spk) + idx4;
  float vx = 0.f, vy = 0.f, vz = 0.f, vw = 0.f;
#pragma unroll 4
  for (int t = 0; t < T_STEPS; ++t) {
    float4 x = p[(size_t)t * BF4];
    float4 s;
    lif_step(vx, x.x, s.x);
    lif_step(vy, x.y, s.y);
    lif_step(vz, x.z, s.z);
    lif_step(vw, x.w, s.w);
    p[(size_t)t * BF4] = s;
  }
}

// ---------------------------------------------------------------------------
extern "C" void kernel_launch(void* const* d_in, const int* in_sizes, int n_in,
                              void* d_out, int out_size, void* d_ws,
                              size_t ws_size, hipStream_t stream) {
  (void)in_sizes; (void)n_in; (void)out_size; (void)ws_size;
  const float* x = (const float*)d_in[0];  // [64,256,1024]
  const float* W = (const float*)d_in[1];  // [1024,1024]
  float* out = (float*)d_out;              // [64,256,1024]

  // Workspace layout: xb (32 MB bf16) | Wb (2 MB bf16)
  bf16* xb = (bf16*)d_ws;
  bf16* Wb = xb + (size_t)M_TOT * FIN;

  {
    const int n4 = (M_TOT * FIN) / 4;   // 4M float4 groups
    cvt_f32_to_bf16<<<n4 / 256, 256, 0, stream>>>(x, xb, n4);
    const int w4 = (FOUT * FIN) / 4;
    cvt_f32_to_bf16<<<w4 / 256, 256, 0, stream>>>(W, Wb, w4);
  }

  dim3 grid(FOUT / BN, M_TOT / BM);     // (8, 128) workgroups
  gemm_bf16_wmma<<<grid, 256, 0, stream>>>(xb, Wb, out);

  lif_scan_kernel<<<(BATCH * FOUT) / (4 * 256), 256, 0, stream>>>(out);
}
